// FactorizedVectorQuantizer_54872502173712
// MI455X (gfx1250) — compile-verified, matrix operand order checked by reading, not run on hardware
//
#include <hip/hip_runtime.h>
#include <math.h>

// ---------------------------------------------------------------------------
// FactorizedVectorQuantizer forward for MI455X (gfx1250), wave32 / WMMA.
//   pass 1: fuse fp32->bf16 conversion (X halves, codebooks), |w|^2, bg flags
//   pass 2: X[T,64] @ W[512,64]^T logits via v_wmma_f32_16x16x32_bf16,
//           LDS tiles staged with GLOBAL_LOAD_ASYNC_TO_LDS_B128 (ASYNCcnt),
//           fused gumbel + restrict_bg argmax on the accumulator layout
//   pass 3: gather w[idx] -> quantized, masked MSE; finalize perplexities
// ---------------------------------------------------------------------------

typedef unsigned short u16;
typedef __attribute__((ext_vector_type(16))) __bf16 v16bf;
typedef __attribute__((ext_vector_type(8)))  float  v8f;

#define CDIM 128
#define HALF 64
#define KSHAPE 512
#define KCOLOR 16
#define NBG 16

union FragU { v16bf v; uint4 q[2]; };

static __device__ inline v16bf mkfrag(uint4 lo, uint4 hi) {
  FragU f; f.q[0] = lo; f.q[1] = hi; return f.v;
}

static __device__ inline u16 f2bf(float f) {
  unsigned u = __float_as_uint(f);
  unsigned r = (u + 0x7fffu + ((u >> 16) & 1u)) >> 16;   // round-nearest-even
  return (u16)r;
}
static __device__ inline unsigned packbf(float a, float b) {
  return (unsigned)f2bf(a) | ((unsigned)f2bf(b) << 16);
}

static __device__ inline unsigned hashu(unsigned x) {
  x ^= x >> 16; x *= 0x7feb352dU;
  x ^= x >> 15; x *= 0x846ca68bU;
  x ^= x >> 16; return x;
}
static __device__ inline float gumbel(unsigned x) {
  float u = ((hashu(x) >> 8) + 0.5f) * (1.0f / 16777216.0f);  // (0,1)
  return -__logf(-__logf(u));
}

// ---- CDNA5 async global -> LDS copy (ASYNCcnt tracked, doc 08) ------------
// VDST VGPR = LDS byte address (low 32 bits of generic shared pointer),
// GVS mode: 64-bit SGPR base + 32-bit per-lane byte offset.
static __device__ inline void async_copy_b128(void* lds_dst, const void* gbase,
                                              unsigned voff) {
  unsigned loff = (unsigned)(unsigned long long)lds_dst;   // LDS offset
  asm volatile("global_load_async_to_lds_b128 %0, %1, %2 offset:0"
               :: "v"(loff), "v"(voff), "s"(gbase) : "memory");
}
static __device__ inline void wait_async_zero() {
  asm volatile("s_wait_asynccnt 0x0" ::: "memory");
}

// --------------------------- init / prep kernels ---------------------------

__global__ void vq_init_kernel(float* usage_s, float* usage_c, float* lossAcc) {
  int t = threadIdx.x;
  if (t < KSHAPE)                usage_s[t] = 0.0f;
  else if (t < KSHAPE + KCOLOR)  usage_c[t - KSHAPE] = 0.0f;
  else if (t == KSHAPE + KCOLOR) *lossAcc = 0.0f;
}

// one thread per token: bg flag + both bf16 halves, single pass over x
__global__ __launch_bounds__(256)
void vq_prep_tokens(const float* __restrict__ x, u16* __restrict__ Xs,
                    u16* __restrict__ Xc, unsigned char* __restrict__ bg, int T) {
  int t = blockIdx.x * blockDim.x + threadIdx.x;
  if (t >= T) return;
  const float4* p = (const float4*)(x + (size_t)t * CDIM);
  __builtin_prefetch(p + 32, 0, 0);                        // global_prefetch_b8
  uint4* ds = (uint4*)(Xs + (size_t)t * HALF);
  uint4* dc = (uint4*)(Xc + (size_t)t * HALF);
  float ab = 0.0f;
#pragma unroll
  for (int k = 0; k < 8; ++k) {                            // first half
    float4 f0 = p[2 * k], f1 = p[2 * k + 1];
    ab += fabsf(f0.x) + fabsf(f0.y) + fabsf(f0.z) + fabsf(f0.w)
        + fabsf(f1.x) + fabsf(f1.y) + fabsf(f1.z) + fabsf(f1.w);
    uint4 o; o.x = packbf(f0.x, f0.y); o.y = packbf(f0.z, f0.w);
    o.z = packbf(f1.x, f1.y); o.w = packbf(f1.z, f1.w);
    ds[k] = o;
  }
#pragma unroll
  for (int k = 0; k < 8; ++k) {                            // second half
    float4 f0 = p[16 + 2 * k], f1 = p[17 + 2 * k];
    ab += fabsf(f0.x) + fabsf(f0.y) + fabsf(f0.z) + fabsf(f0.w)
        + fabsf(f1.x) + fabsf(f1.y) + fabsf(f1.z) + fabsf(f1.w);
    uint4 o; o.x = packbf(f0.x, f0.y); o.y = packbf(f0.z, f0.w);
    o.z = packbf(f1.x, f1.y); o.w = packbf(f1.z, f1.w);
    dc[k] = o;
  }
  bg[t] = (ab < 1e-6f) ? 1 : 0;
}

// one thread per code: |w|^2 + bf16 conversion of both codebooks
__global__ void vq_prep_codes(const float* __restrict__ ws_, const float* __restrict__ wc_,
                              u16* __restrict__ Ws, u16* __restrict__ Wc,
                              float* __restrict__ wn_s, float* __restrict__ wn_c) {
  int k = blockIdx.x * blockDim.x + threadIdx.x;
  const float4* src; uint4* dst; float* o;
  if (k < KSHAPE) {
    src = (const float4*)(ws_ + (size_t)k * HALF);
    dst = (uint4*)(Ws + (size_t)k * HALF); o = &wn_s[k];
  } else if (k < KSHAPE + KCOLOR) {
    int kk = k - KSHAPE;
    src = (const float4*)(wc_ + (size_t)kk * HALF);
    dst = (uint4*)(Wc + (size_t)kk * HALF); o = &wn_c[kk];
  } else return;
  float s = 0.0f;
#pragma unroll
  for (int q = 0; q < 8; ++q) {
    float4 f0 = src[2 * q], f1 = src[2 * q + 1];
    s += f0.x * f0.x + f0.y * f0.y + f0.z * f0.z + f0.w * f0.w
       + f1.x * f1.x + f1.y * f1.y + f1.z * f1.z + f1.w * f1.w;
    uint4 ou; ou.x = packbf(f0.x, f0.y); ou.y = packbf(f0.z, f0.w);
    ou.z = packbf(f1.x, f1.y); ou.w = packbf(f1.z, f1.w);
    dst[q] = ou;
  }
  *o = s;
}

// ------------------- shape: WMMA logits + gumbel argmax --------------------
// Block = 256 threads (8 waves), 128 tokens per block, codes in two halves
// of 256; all LDS tiles filled by async b128 copies (no VALU in the copy).

__global__ __launch_bounds__(256)
void vq_shape_kernel(const u16* __restrict__ Xs, const u16* __restrict__ Ws,
                     const float* __restrict__ wnorm,
                     const unsigned char* __restrict__ bg,
                     const unsigned char* __restrict__ valid,
                     int* __restrict__ sidx, float* __restrict__ usage, int T) {
  __shared__ u16           ldsW[256 * HALF];   // 32 KB: half of the codebook
  __shared__ u16           ldsX[128 * HALF];   // 16 KB: token tile
  __shared__ float         ldsWn[256];
  __shared__ unsigned char ldsBg[128];

  const int tid  = threadIdx.x;
  const int lane = tid & 31;
  const int wv   = tid >> 5;                   // wave 0..7 -> 16 tokens each
  const int hi   = lane >> 4;
  const int lo16 = lane & 15;
  const int tokBase = blockIdx.x * 128;

  // ---- async-stage X tile: 1024 x 16B chunks ----
  {
    const u16* xbase = Xs + (size_t)tokBase * HALF;
#pragma unroll
    for (int c = tid; c < 1024; c += 256)
      async_copy_b128(&ldsX[c * 8], xbase, (unsigned)c * 16u);
    if (tid < 128) ldsBg[tid] = bg[tokBase + tid];
  }

  v16bf a0 = {}, a1 = {};
  float best[8]; int bidx[8];
#pragma unroll
  for (int r = 0; r < 8; ++r) { best[r] = -INFINITY; bidx[r] = 0; }

  for (int h = 0; h < 2; ++h) {
    __syncthreads();                           // h=1: ldsW fully consumed
    // ---- async-stage 256 codebook rows: 2048 x 16B chunks ----
    {
      const u16* wbase = Ws + (size_t)(h * 256) * HALF;
#pragma unroll
      for (int c = tid; c < 2048; c += 256)
        async_copy_b128(&ldsW[c * 8], wbase, (unsigned)c * 16u);
      ldsWn[tid] = wnorm[h * 256 + tid];
    }
    wait_async_zero();                         // this wave's copies done
    __syncthreads();                           // all waves' copies visible

    if (h == 0) {                              // A frags: reused for both halves
      int row = wv * 16 + lo16;
      const uint4* p = (const uint4*)&ldsX[row * HALF];
      a0 = mkfrag(p[0 + hi], p[2 + hi]);       // K 0..31
      a1 = mkfrag(p[4 + hi], p[6 + hi]);       // K 32..63
    }

    for (int nt = 0; nt < 16; ++nt) {
      v16bf b0, b1;
      {
        int row = nt * 16 + lo16;              // code row acts as B column
        const uint4* p = (const uint4*)&ldsW[row * HALF];
        b0 = mkfrag(p[0 + hi], p[2 + hi]);
        b1 = mkfrag(p[4 + hi], p[6 + hi]);
      }
      v8f acc = {};
      acc = __builtin_amdgcn_wmma_f32_16x16x32_bf16(false, a0, false, b0,
                                                    (short)0, acc, false, false);
      acc = __builtin_amdgcn_wmma_f32_16x16x32_bf16(false, a1, false, b1,
                                                    (short)0, acc, false, false);

      int   code     = h * 256 + nt * 16 + lo16;
      float wn       = ldsWn[nt * 16 + lo16];
      bool  codeIsBg = (code < NBG);
#pragma unroll
      for (int r = 0; r < 8; ++r) {
        int rowLocal = wv * 16 + (hi ? 8 + r : r);
        int tok      = tokBase + rowLocal;
        bool rowBg   = (ldsBg[rowLocal] != 0);
        // argmax(-(|x|^2+|w|^2-2xw)+g) == argmax(2xw - |w|^2 + g)
        float s = 2.0f * acc[r] - wn + gumbel((unsigned)tok * 0x9E3779B9u
                                              ^ (unsigned)code * 0x85EBCA6Bu
                                              ^ 0x5151AEDBu);
        if (rowBg != codeIsBg) s = -INFINITY;  // restrict_bg masking
        if (s > best[r]) { best[r] = s; bidx[r] = code; }
      }
    }
  }

  // ---- reduce argmax across the 16 lanes holding one row ----
#pragma unroll
  for (int m = 1; m < 16; m <<= 1) {
#pragma unroll
    for (int r = 0; r < 8; ++r) {
      float ob = __shfl_xor(best[r], m, 32);
      int   oi = __shfl_xor(bidx[r], m, 32);
      if (ob > best[r] || (ob == best[r] && oi < bidx[r])) { best[r] = ob; bidx[r] = oi; }
    }
  }
  if (lo16 == 0) {
#pragma unroll
    for (int r = 0; r < 8; ++r) {
      int tok = tokBase + wv * 16 + (hi ? 8 + r : r);
      sidx[tok] = bidx[r];
      atomicAdd(&usage[bidx[r]], valid[tok] ? 1.0f : 0.0f);
    }
  }
}

// ------------------- color: single 16-code tile, same scheme ---------------

__global__ __launch_bounds__(256)
void vq_color_kernel(const u16* __restrict__ Xc, const u16* __restrict__ Wc,
                     const float* __restrict__ wnorm,
                     const unsigned char* __restrict__ valid,
                     int* __restrict__ cidx, float* __restrict__ usage, int T) {
  __shared__ u16   ldsX[128 * HALF];           // 16 KB
  __shared__ u16   ldsW[KCOLOR * HALF];        // 2 KB
  __shared__ float ldsWn[KCOLOR];

  const int tid  = threadIdx.x;
  const int lane = tid & 31;
  const int wv   = tid >> 5;
  const int hi   = lane >> 4;
  const int lo16 = lane & 15;
  const int tokBase = blockIdx.x * 128;

  {
    const u16* xbase = Xc + (size_t)tokBase * HALF;
#pragma unroll
    for (int c = tid; c < 1024; c += 256)
      async_copy_b128(&ldsX[c * 8], xbase, (unsigned)c * 16u);
    if (tid < 128) async_copy_b128(&ldsW[tid * 8], Wc, (unsigned)tid * 16u);
    if (tid < KCOLOR) ldsWn[tid] = wnorm[tid];
  }
  wait_async_zero();
  __syncthreads();

  v16bf a0, a1, b0, b1;
  {
    int row = wv * 16 + lo16;
    const uint4* p = (const uint4*)&ldsX[row * HALF];
    a0 = mkfrag(p[0 + hi], p[2 + hi]);
    a1 = mkfrag(p[4 + hi], p[6 + hi]);
    const uint4* q = (const uint4*)&ldsW[lo16 * HALF];
    b0 = mkfrag(q[0 + hi], q[2 + hi]);
    b1 = mkfrag(q[4 + hi], q[6 + hi]);
  }
  v8f acc = {};
  acc = __builtin_amdgcn_wmma_f32_16x16x32_bf16(false, a0, false, b0,
                                                (short)0, acc, false, false);
  acc = __builtin_amdgcn_wmma_f32_16x16x32_bf16(false, a1, false, b1,
                                                (short)0, acc, false, false);

  float best[8]; int bidx[8];
  float wn = ldsWn[lo16];
#pragma unroll
  for (int r = 0; r < 8; ++r) {
    int tok = tokBase + wv * 16 + (hi ? 8 + r : r);
    best[r] = 2.0f * acc[r] - wn + gumbel((unsigned)tok * 0x27d4eb2fu
                                          ^ (unsigned)lo16 * 0x9E3779B9u
                                          ^ 0xC0105EEDu);
    bidx[r] = lo16;
  }
#pragma unroll
  for (int m = 1; m < 16; m <<= 1) {
#pragma unroll
    for (int r = 0; r < 8; ++r) {
      float ob = __shfl_xor(best[r], m, 32);
      int   oi = __shfl_xor(bidx[r], m, 32);
      if (ob > best[r] || (ob == best[r] && oi < bidx[r])) { best[r] = ob; bidx[r] = oi; }
    }
  }
  if (lo16 == 0) {
#pragma unroll
    for (int r = 0; r < 8; ++r) {
      int tok = tokBase + wv * 16 + (hi ? 8 + r : r);
      cidx[tok] = bidx[r];
      atomicAdd(&usage[bidx[r]], valid[tok] ? 1.0f : 0.0f);
    }
  }
}

// ---------------- gather w[idx] -> quantized, masked MSE -------------------

__global__ __launch_bounds__(256)
void vq_gather_loss_kernel(const float* __restrict__ x,
                           const float* __restrict__ ws, const float* __restrict__ wc,
                           const int* __restrict__ sidx, const int* __restrict__ cidx,
                           const unsigned char* __restrict__ valid,
                           float* __restrict__ outQ, float* __restrict__ outIdx,
                           float* __restrict__ lossAcc, int T) {
  __shared__ float red[256];
  int gid = blockIdx.x * 256 + threadIdx.x;
  int t = gid >> 5;                            // 32 float4 chunks per token
  int j = gid & 31;
  float4 q;
  if (j < 16) { int si = sidx[t]; q = *(const float4*)(ws + (size_t)si * HALF + j * 4); }
  else        { int ci = cidx[t]; q = *(const float4*)(wc + (size_t)ci * HALF + (j - 16) * 4); }
  float4 xv = *(const float4*)(x + (size_t)t * CDIM + j * 4);
  *(float4*)(outQ + (size_t)t * CDIM + j * 4) = q;

  float m = valid[t] ? 1.0f : 0.0f;
  float d0 = q.x - xv.x, d1 = q.y - xv.y, d2 = q.z - xv.z, d3 = q.w - xv.w;
  red[threadIdx.x] = (d0 * d0 + d1 * d1 + d2 * d2 + d3 * d3) * m;
  __syncthreads();
  for (int st = 128; st > 0; st >>= 1) {
    if (threadIdx.x < st) red[threadIdx.x] += red[threadIdx.x + st];
    __syncthreads();
  }
  if (threadIdx.x == 0) atomicAdd(lossAcc, red[0]);
  if (j == 0) outIdx[t] = (float)sidx[t];
}

// ------------------------------ finalize -----------------------------------

__global__ __launch_bounds__(512)
void vq_final_kernel(const float* __restrict__ usage_s, const float* __restrict__ usage_c,
                     const float* __restrict__ lossAcc, float* __restrict__ out,
                     size_t qElems, int T) {
  __shared__ float red[512];
  int tid = threadIdx.x;

  red[tid] = usage_s[tid];                      // vcount = sum(shape usage)
  __syncthreads();
  for (int st = 256; st > 0; st >>= 1) {
    if (tid < st) red[tid] += red[tid + st];
    __syncthreads();
  }
  float vcount = red[0];
  __syncthreads();

  float p = usage_s[tid] / vcount;
  red[tid] = p * __logf(p + 1e-10f);
  __syncthreads();
  for (int st = 256; st > 0; st >>= 1) {
    if (tid < st) red[tid] += red[tid + st];
    __syncthreads();
  }
  float perpS = __expf(-red[0]);
  __syncthreads();

  float ec = 0.0f;
  if (tid < KCOLOR) { float pc = usage_c[tid] / vcount; ec = pc * __logf(pc + 1e-10f); }
  red[tid] = ec;
  __syncthreads();
  for (int st = 256; st > 0; st >>= 1) {
    if (tid < st) red[tid] += red[tid + st];
    __syncthreads();
  }
  float perpC = __expf(-red[0]);

  if (tid == 0) {
    float loss = 1.25f * (*lossAcc) / (vcount * (float)CDIM);
    out[qElems + 0] = loss;
    out[qElems + 1] = perpS;
    out[qElems + 2] = perpC;
  }
  float* outUs = out + qElems + 3 + (size_t)T;
  outUs[tid] = usage_s[tid];
  if (tid < KCOLOR) outUs[KSHAPE + tid] = usage_c[tid];
}

// ------------------------------ launcher -----------------------------------

extern "C" void kernel_launch(void* const* d_in, const int* in_sizes, int n_in,
                              void* d_out, int out_size, void* d_ws, size_t ws_size,
                              hipStream_t stream) {
  const float*         x     = (const float*)d_in[0];
  const unsigned char* valid = (const unsigned char*)d_in[1];   // bool mask
  const float*         w_s   = (const float*)d_in[2];
  const float*         w_c   = (const float*)d_in[3];
  float* out = (float*)d_out;

  const int    T      = in_sizes[0] / CDIM;     // 65536
  const size_t qElems = (size_t)T * CDIM;

  // workspace carve-out (256B aligned sections)
  char* p = (char*)d_ws;
  auto carve = [&](size_t bytes) {
    char* r = p;
    p += (bytes + 255) & ~(size_t)255;
    return r;
  };
  float* usage_s = (float*)carve(KSHAPE * sizeof(float));
  float* usage_c = (float*)carve(KCOLOR * sizeof(float));
  float* wn_s    = (float*)carve(KSHAPE * sizeof(float));
  float* wn_c    = (float*)carve(KCOLOR * sizeof(float));
  float* lossAcc = (float*)carve(sizeof(float));
  int*   sidx    = (int*)carve((size_t)T * sizeof(int));
  int*   cidx    = (int*)carve((size_t)T * sizeof(int));
  unsigned char* bgf = (unsigned char*)carve((size_t)T);
  u16* Xs = (u16*)carve((size_t)T * HALF * sizeof(u16));     // bf16 halves
  u16* Xc = (u16*)carve((size_t)T * HALF * sizeof(u16));
  u16* Ws = (u16*)carve((size_t)KSHAPE * HALF * sizeof(u16));
  u16* Wc = (u16*)carve((size_t)KCOLOR * HALF * sizeof(u16));

  vq_init_kernel<<<1, 544, 0, stream>>>(usage_s, usage_c, lossAcc);
  vq_prep_tokens<<<(T + 255) / 256, 256, 0, stream>>>(x, Xs, Xc, bgf, T);
  vq_prep_codes<<<(KSHAPE + KCOLOR + 255) / 256, 256, 0, stream>>>(
      w_s, w_c, Ws, Wc, wn_s, wn_c);
  vq_shape_kernel<<<T / 128, 256, 0, stream>>>(Xs, Ws, wn_s, bgf, valid,
                                               sidx, usage_s, T);
  vq_color_kernel<<<T / 128, 256, 0, stream>>>(Xc, Wc, wn_c, valid,
                                               cidx, usage_c, T);
  vq_gather_loss_kernel<<<(T * 32) / 256, 256, 0, stream>>>(
      x, w_s, w_c, sidx, cidx, valid, out, out + qElems + 3, lossAcc, T);
  vq_final_kernel<<<1, 512, 0, stream>>>(usage_s, usage_c, lossAcc, out, qElems, T);
}
